// GraphSAGE_33397665694044
// MI455X (gfx1250) — matrix operand order repaired and checked by reading, not compiled, not run on hardware
//
#include <hip/hip_runtime.h>

// ---------------- constants ----------------
constexpr int   NN     = 100000;   // nodes
constexpr int   EE     = 1600000;  // edges
constexpr int   CC     = 128;      // channels
constexpr int   NSTRIP = NN / 16;  // 6250 row strips
constexpr float BN_EPS = 1e-5f;

typedef __attribute__((ext_vector_type(16))) _Float16 v16h;
typedef __attribute__((ext_vector_type(8)))  _Float16 v8h;
typedef __attribute__((ext_vector_type(8)))  float    v8f;
typedef __attribute__((ext_vector_type(4)))  int      v4i;

#if __has_builtin(__builtin_amdgcn_global_load_async_to_lds_b128)
#define HAVE_ASYNC_LDS 1
#else
#define HAVE_ASYNC_LDS 0
#endif

// ---------------- weight pre-swizzle ----------------
// WMMA B (32x16 f16, wave32): lane owns column n = tn*16 + (lane&15), g = lane>>4;
// its 16 halves are K = c*32 + g*8 + j (j<8) and K = c*32 + 16 + g*8 + (j-8) (j>=8) for K-chunk c.
// Store split into two 16B half-blocks per lane for conflict-free ds_load_b128:
//   dst[ ((tile*2 + h)*32 + lane)*8 + j8 ]  with tile = tn*4 + c, K = c*32 + h*16 + g*8 + j8.
__global__ void k_convert_swz_b(const float* __restrict__ src, _Float16* __restrict__ dst) {
    int tid = blockIdx.x * blockDim.x + threadIdx.x;   // [0, 16384)
    if (tid >= CC * CC) return;
    int j8   = tid & 7;
    int lane = (tid >> 3) & 31;
    int h    = (tid >> 8) & 1;
    int c    = (tid >> 9) & 3;
    int tn   = (tid >> 11) & 7;
    int g = lane >> 4;
    int n = tn * 16 + (lane & 15);
    int k = c * 32 + h * 16 + g * 8 + j8;
    dst[tid] = (_Float16)src[k * CC + n];
}

// ---------------- LDS staging (async-to-LDS when available) ----------------
__device__ __forceinline__ void stage_weights(const _Float16* __restrict__ g,
                                              _Float16* s, int halves) {
    int tid = threadIdx.x, nt = blockDim.x;
#if HAVE_ASYNC_LDS
    for (int i = tid; i < (halves >> 3); i += nt) {
        auto gsrc = (__attribute__((address_space(1))) v4i*)
                        (unsigned long long)(g + i * 8);
        auto ldst = (__attribute__((address_space(3))) v4i*)
                        (unsigned)(unsigned long long)(s + i * 8);
        __builtin_amdgcn_global_load_async_to_lds_b128(gsrc, ldst, 0, 0);
    }
#if __has_builtin(__builtin_amdgcn_s_wait_asynccnt)
    __builtin_amdgcn_s_wait_asynccnt(0);
#else
    asm volatile("s_wait_asynccnt 0" ::: "memory");
#endif
#else
    for (int i = tid; i < (halves >> 3); i += nt)
        *(v8h*)(s + i * 8) = *(const v8h*)(g + i * 8);
#endif
    __syncthreads();
}

// ---------------- graph kernels ----------------

__global__ void k_degree(const int* __restrict__ dsti, float* __restrict__ deg) {
    int e = blockIdx.x * blockDim.x + threadIdx.x;
    if (e < EE) atomicAdd(&deg[dsti[e]], 1.0f);
}

__global__ void k_invdeg(float* __restrict__ deg) {
    int i = blockIdx.x * blockDim.x + threadIdx.x;
    if (i < NN) deg[i] = 1.0f / fmaxf(deg[i], 1.0f);
}

__global__ void k_scatter3(const float* __restrict__ x, const int* __restrict__ srci,
                           const int* __restrict__ dsti, float* __restrict__ agg) {
    int e = blockIdx.x * blockDim.x + threadIdx.x;
    if (e >= EE) return;
    int s = srci[e], d = dsti[e];
    atomicAdd(&agg[d * 3 + 0], x[s * 3 + 0]);
    atomicAdd(&agg[d * 3 + 1], x[s * 3 + 1]);
    atomicAdd(&agg[d * 3 + 2], x[s * 3 + 2]);
}

// f16 message gather (16 threads/edge, 8 halves = 16B load), f32 atomic accumulate.
__global__ void k_scatter128h(const _Float16* __restrict__ x16, const int* __restrict__ srci,
                              const int* __restrict__ dsti, float* __restrict__ agg) {
    long t = (long)blockIdx.x * blockDim.x + threadIdx.x;
    long e = t >> 4;
    if (e >= EE) return;
    int c8 = ((int)t & 15) * 8;
    int s = srci[e], d = dsti[e];
    const v8h v = *(const v8h*)(x16 + (long)s * CC + c8);
    float* ap = agg + (long)d * CC + c8;
#pragma unroll
    for (int i = 0; i < 8; ++i) atomicAdd(ap + i, (float)v[i]);
}

// agg(f32) * invdeg -> row-major f16 A buffer (once per layer)
__global__ void k_prep_a(const float* __restrict__ agg, const float* __restrict__ invdeg,
                         _Float16* __restrict__ a16) {
    long t = (long)blockIdx.x * blockDim.x + threadIdx.x;
    if (t >= (long)NN * CC) return;
    int row = (int)(t >> 7);
    a16[t] = (_Float16)(agg[t] * invdeg[row]);
}

// Layer 1: Cin=3, plain VALU; writes row-major f16 activations.
__global__ void k_layer1(const float* __restrict__ pos, const float* __restrict__ agg,
                         const float* __restrict__ invdeg,
                         const float* __restrict__ wl, const float* __restrict__ bl,
                         const float* __restrict__ wr,
                         const float* __restrict__ gg, const float* __restrict__ be,
                         const float* __restrict__ mu, const float* __restrict__ vv,
                         _Float16* __restrict__ out16) {
    int node = blockIdx.x;
    int ch   = threadIdx.x;
    float id = invdeg[node];
    float a0 = agg[node * 3 + 0] * id, a1 = agg[node * 3 + 1] * id, a2 = agg[node * 3 + 2] * id;
    float x0 = pos[node * 3 + 0], x1 = pos[node * 3 + 1], x2 = pos[node * 3 + 2];
    float val = a0 * wl[0 * CC + ch] + a1 * wl[1 * CC + ch] + a2 * wl[2 * CC + ch] + bl[ch]
              + x0 * wr[0 * CC + ch] + x1 * wr[1 * CC + ch] + x2 * wr[2 * CC + ch];
    float sc = gg[ch] * rsqrtf(vv[ch] + BN_EPS);
    val = (val - mu[ch]) * sc + be[ch];
    out16[(long)node * CC + ch] = (_Float16)fmaxf(val, 0.0f);
}

// ---------------- WMMA helpers ----------------
// A 16x32 f16 from ROW-MAJOR f16: lane row = tile*16 + (lane&15), g = lane>>4;
// halves = row16[kk + g*8 .. +8) ++ row16[kk + g*8 + 16 .. +8)  (two 16B runs).
__device__ __forceinline__ v16h load_a16(const _Float16* __restrict__ row16, int kk, int g) {
    const v8h lo = *(const v8h*)(row16 + kk + g * 8);
    const v8h hi = *(const v8h*)(row16 + kk + g * 8 + 16);
    v16h a;
#pragma unroll
    for (int i = 0; i < 8; ++i) { a[i] = lo[i]; a[8 + i] = hi[i]; }
    return a;
}

// B tile from LDS (pre-swizzled): two ds_load_b128, 16B lane stride (conflict-free).
__device__ __forceinline__ v16h load_b_lds(const _Float16* s, int tn, int c, int lane) {
    int tile = tn * 4 + c;
    const v8h lo = *(const v8h*)(s + ((tile * 2 + 0) * 32 + lane) * 8);
    const v8h hi = *(const v8h*)(s + ((tile * 2 + 1) * 32 + lane) * 8);
    v16h b;
#pragma unroll
    for (int i = 0; i < 8; ++i) { b[i] = lo[i]; b[8 + i] = hi[i]; }
    return b;
}

// SAGE layer: 8 waves/block, each wave computes a 16x128 row strip (8 accumulators,
// 64 WMMAs). Wl|Wr (contiguous, 64KB) staged in LDS via async-to-LDS.
__global__ void __launch_bounds__(256)
k_sage_wmma(const _Float16* __restrict__ X16, const _Float16* __restrict__ A16,
            const _Float16* __restrict__ WLR,   // swizzled Wl followed by Wr
            const float* __restrict__ BL,
            const float* __restrict__ GG, const float* __restrict__ BE,
            const float* __restrict__ MU, const float* __restrict__ VV,
            _Float16* __restrict__ OUT16) {
    __shared__ _Float16 sW[2 * CC * CC];
    stage_weights(WLR, sW, 2 * CC * CC);
    const _Float16* sWL = sW;
    const _Float16* sWR = sW + CC * CC;

    int tm = blockIdx.x * (blockDim.x >> 5) + (threadIdx.x >> 5);
    if (tm >= NSTRIP) return;                      // wave-uniform (after barrier)
    int lane = threadIdx.x & 31;
    int g2 = lane >> 4, hm = lane & 15;
    const _Float16* xrow = X16 + (long)(tm * 16 + hm) * CC;
    const _Float16* arow = A16 + (long)(tm * 16 + hm) * CC;

    v8f acc[8] = {};
#pragma unroll
    for (int c = 0; c < 4; ++c) {
        v16h aA = load_a16(arow, c * 32, g2);
        v16h aX = load_a16(xrow, c * 32, g2);
        v16h bL[8];
#pragma unroll
        for (int tn = 0; tn < 8; ++tn) bL[tn] = load_b_lds(sWL, tn, c, lane);
#pragma unroll
        for (int tn = 0; tn < 8; ++tn)
            acc[tn] = __builtin_amdgcn_wmma_f32_16x16x32_f16(false, aA, false, bL[tn], (short)0, acc[tn], false, false);
        v16h bR[8];
#pragma unroll
        for (int tn = 0; tn < 8; ++tn) bR[tn] = load_b_lds(sWR, tn, c, lane);
#pragma unroll
        for (int tn = 0; tn < 8; ++tn)
            acc[tn] = __builtin_amdgcn_wmma_f32_16x16x32_f16(false, aX, false, bR[tn], (short)0, acc[tn], false, false);
    }
#pragma unroll
    for (int tn = 0; tn < 8; ++tn) {
        int col = tn * 16 + hm;                    // this lane's output channel
        float bl = BL[col];
        float sc = GG[col] * rsqrtf(VV[col] + BN_EPS);
        float mu = MU[col], sh = BE[col];
#pragma unroll
        for (int r = 0; r < 8; ++r) {
            int mrow = tm * 16 + r + 8 * g2;       // C/D layout: acc[r] = (M = r+8g, N = col)
            float v = (acc[tn][r] + bl - mu) * sc + sh;
            OUT16[(long)mrow * CC + col] = (_Float16)fmaxf(v, 0.0f);
        }
    }
}

// FC layer: 16x128 strip per wave; optional ReLU; writes f16 (chain) and/or f32 (final).
__global__ void __launch_bounds__(256)
k_fc_wmma(const _Float16* __restrict__ X16, const _Float16* __restrict__ W,
          const float* __restrict__ B,
          _Float16* __restrict__ OUT16, float* __restrict__ OUTF, int relu) {
    __shared__ _Float16 sW[CC * CC];
    stage_weights(W, sW, CC * CC);

    int tm = blockIdx.x * (blockDim.x >> 5) + (threadIdx.x >> 5);
    if (tm >= NSTRIP) return;
    int lane = threadIdx.x & 31;
    int g2 = lane >> 4, hm = lane & 15;
    const _Float16* xrow = X16 + (long)(tm * 16 + hm) * CC;

    v8f acc[8] = {};
#pragma unroll
    for (int c = 0; c < 4; ++c) {
        v16h a = load_a16(xrow, c * 32, g2);
        v16h b[8];
#pragma unroll
        for (int tn = 0; tn < 8; ++tn) b[tn] = load_b_lds(sW, tn, c, lane);
#pragma unroll
        for (int tn = 0; tn < 8; ++tn)
            acc[tn] = __builtin_amdgcn_wmma_f32_16x16x32_f16(false, a, false, b[tn], (short)0, acc[tn], false, false);
    }
#pragma unroll
    for (int tn = 0; tn < 8; ++tn) {
        int col = tn * 16 + hm;
        float bias = B[col];
#pragma unroll
        for (int r = 0; r < 8; ++r) {
            int mrow = tm * 16 + r + 8 * g2;
            float v = acc[tn][r] + bias;
            if (relu) v = fmaxf(v, 0.0f);
            if (OUT16) OUT16[(long)mrow * CC + col] = (_Float16)v;
            if (OUTF)  OUTF[(long)mrow * CC + col] = v;
        }
    }
}

// ---------------- host launch ----------------

extern "C" void kernel_launch(void* const* d_in, const int* in_sizes, int n_in,
                              void* d_out, int out_size, void* d_ws, size_t ws_size,
                              hipStream_t stream) {
    (void)in_sizes; (void)n_in; (void)out_size; (void)ws_size;

    const float* pos = (const float*)d_in[0];
    const int*   ei  = (const int*)d_in[1];
    const int* srci = ei;
    const int* dsti = ei + EE;

    // dict order: pos, edge_index, {wl,bl,wr,g,be,m,v} x4 (bases 2,9,16,23), wf1,bf1,wf2,bf2,wf3,bf3 (30..35)
    char* ws = (char*)d_ws;
    auto alignup = [](size_t x) { return (x + 255) & ~(size_t)255; };
    size_t off = 0;
    float*    deg  = (float*)(ws + off);    off += alignup((size_t)NN * 4);
    float*    agg  = (float*)(ws + off);    off += alignup((size_t)NN * CC * 4);
    _Float16* a16  = (_Float16*)(ws + off); off += alignup((size_t)NN * CC * 2);   // agg*invdeg (f16)
    _Float16* x16a = (_Float16*)(ws + off); off += alignup((size_t)NN * CC * 2);   // ping
    _Float16* x16b = (_Float16*)(ws + off); off += alignup((size_t)NN * CC * 2);   // pong
    _Float16* w16  = (_Float16*)(ws + off);                                        // 9 swizzled weights

    // pre-swizzle the nine 128x128 weights: wl2,wr2,wl3,wr3,wl4,wr4,wf1,wf2,wf3
    const int widx[9] = {9, 11, 16, 18, 23, 25, 30, 32, 34};
    for (int i = 0; i < 9; ++i) {
        k_convert_swz_b<<<(CC * CC + 255) / 256, 256, 0, stream>>>(
            (const float*)d_in[widx[i]], w16 + (size_t)i * CC * CC);
    }

    // degrees -> invdeg (stored in deg)
    (void)hipMemsetAsync(deg, 0, (size_t)NN * 4, stream);
    k_degree<<<(EE + 255) / 256, 256, 0, stream>>>(dsti, deg);
    k_invdeg<<<(NN + 255) / 256, 256, 0, stream>>>(deg);

    // ---- layer 1 (Cin=3, f32 gather from pos) ----
    (void)hipMemsetAsync(agg, 0, (size_t)NN * 3 * 4, stream);
    k_scatter3<<<(EE + 255) / 256, 256, 0, stream>>>(pos, srci, dsti, agg);
    k_layer1<<<NN, CC, 0, stream>>>(pos, agg, deg,
                                    (const float*)d_in[2], (const float*)d_in[3], (const float*)d_in[4],
                                    (const float*)d_in[5], (const float*)d_in[6], (const float*)d_in[7],
                                    (const float*)d_in[8], x16a);

    // ---- layers 2..4 (WMMA, f16 activations end-to-end) ----
    const int gBlocks  = (NSTRIP + 7) / 8;       // 8 waves (256 thr) per block
    const int scBlk128 = (int)(((long)EE * 16 + 255) / 256);
    const int prepBlk  = (int)(((long)NN * CC + 255) / 256);

    _Float16* xin = x16a;
    _Float16* xout = x16b;
    const int base[3] = {9, 16, 23};
    for (int L = 0; L < 3; ++L) {
        int b = base[L];
        (void)hipMemsetAsync(agg, 0, (size_t)NN * CC * 4, stream);
        k_scatter128h<<<scBlk128, 256, 0, stream>>>(xin, srci, dsti, agg);
        k_prep_a<<<prepBlk, 256, 0, stream>>>(agg, deg, a16);
        k_sage_wmma<<<gBlocks, 256, 0, stream>>>(
            xin, a16,
            w16 + (size_t)(2 * L) * CC * CC,       // Wl|Wr contiguous (swizzled)
            (const float*)d_in[b + 1],             // bl
            (const float*)d_in[b + 3],             // gamma
            (const float*)d_in[b + 4],             // beta
            (const float*)d_in[b + 5],             // mean
            (const float*)d_in[b + 6],             // var
            xout);
        _Float16* t = xin; xin = xout; xout = t;
    }
    // xin holds layer-4 activations (f16)

    // ---- FC head: relu only between fc1 and fc2; fc3 writes f32 d_out ----
    k_fc_wmma<<<gBlocks, 256, 0, stream>>>(xin,  w16 + (size_t)6 * CC * CC, (const float*)d_in[31], xout, nullptr, 1);
    k_fc_wmma<<<gBlocks, 256, 0, stream>>>(xout, w16 + (size_t)7 * CC * CC, (const float*)d_in[33], xin,  nullptr, 0);
    k_fc_wmma<<<gBlocks, 256, 0, stream>>>(xin,  w16 + (size_t)8 * CC * CC, (const float*)d_in[35], nullptr, (float*)d_out, 0);
}
